// Attention_32495722562177
// MI455X (gfx1250) — compile-verified, matrix operand order checked by reading, not run on hardware
//
#include <hip/hip_runtime.h>

// ---------------------------------------------------------------------------
// Causal MHA forward, fixed shape B=2, T=2048, E=1024, H=16 (d=64), gfx1250.
//   1) cvt x,y -> bf16 ; transpose+cvt Wq/Wk/Wv/Wo -> bf16 [N][K]
//   2) WMMA GEMM (bf16 in, f32 acc, async-LDS double-buffered): Q/K/V
//      (Q epilogue folds the 1/sqrt(d)=0.125 softmax scale)
//   3) flash attention, KT=64 tiles; interior tiles take a mask-free path
//   4) WMMA GEMM: out = attn @ Wo + bo -> f32 [B,T,E]
// CDNA5 paths: v_wmma_f32_16x16x32_bf16, global_load_async_to_lds_b128 +
// s_wait_asynccnt, ds_swizzle xor-reductions, s_wait_dscnt wave-local LDS.
// ---------------------------------------------------------------------------

typedef __bf16 bf16_t;
typedef __attribute__((ext_vector_type(16))) __bf16 v16bf;
typedef __attribute__((ext_vector_type(8)))  float  v8f;

template <bool V> struct BoolC { static constexpr bool value = V; };

#define WMMA_BF16(a, b, c) \
  __builtin_amdgcn_wmma_f32_16x16x32_bf16(false, (a), false, (b), (short)0, (c), false, false)

// Async global->LDS copy of 16 bytes (ASYNCcnt-tracked DMA path).
// Generic pointers to __shared__ carry the LDS byte offset in the low 32 bits.
__device__ __forceinline__ void async_copy_b128(const bf16_t* gsrc, bf16_t* ldst) {
  unsigned laddr = (unsigned)(uintptr_t)ldst;
  asm volatile("global_load_async_to_lds_b128 %0, %1, off"
               :: "v"(laddr), "v"(gsrc) : "memory");
}
__device__ __forceinline__ void wait_async() {
  asm volatile("s_wait_asynccnt 0x0" ::: "memory");
}

// 16x32 bf16 fragment from a row-major, K-contiguous buffer.
// CDNA5 16-bit A layout: lane (r=lane&15) holds row/col r; lanes 0-15 carry
// K 0..7 / 16..23, lanes 16-31 carry K 8..15 / 24..31 -> two b128 loads.
// B uses the mirrored layout (lane indexes N), so one loader serves both.
__device__ __forceinline__ v16bf load_frag(const bf16_t* base, int ld) {
  const int lane = threadIdx.x & 31;
  const int r    = lane & 15;
  const int hk   = lane >> 4;
  const bf16_t* p = base + (size_t)r * ld;
  union { v16bf v; uint4 u[2]; } f;
  f.u[0] = *(const uint4*)(p + hk * 8);
  f.u[1] = *(const uint4*)(p + 16 + hk * 8);
  return f.v;
}

// 16-lane xor reductions via DS_SWIZZLE_B32 (and=0x1F, xor=1/2/4/8).
__device__ __forceinline__ float red_max16(float x) {
  x = fmaxf(x, __int_as_float(__builtin_amdgcn_ds_swizzle(__float_as_int(x), 0x041F)));
  x = fmaxf(x, __int_as_float(__builtin_amdgcn_ds_swizzle(__float_as_int(x), 0x081F)));
  x = fmaxf(x, __int_as_float(__builtin_amdgcn_ds_swizzle(__float_as_int(x), 0x101F)));
  x = fmaxf(x, __int_as_float(__builtin_amdgcn_ds_swizzle(__float_as_int(x), 0x201F)));
  return x;
}
__device__ __forceinline__ float red_sum16(float x) {
  x += __int_as_float(__builtin_amdgcn_ds_swizzle(__float_as_int(x), 0x041F));
  x += __int_as_float(__builtin_amdgcn_ds_swizzle(__float_as_int(x), 0x081F));
  x += __int_as_float(__builtin_amdgcn_ds_swizzle(__float_as_int(x), 0x101F));
  x += __int_as_float(__builtin_amdgcn_ds_swizzle(__float_as_int(x), 0x201F));
  return x;
}

// ---------------------------------------------------------------------------
__global__ void cvt_f32_bf16(const float* __restrict__ in, bf16_t* __restrict__ out, size_t n) {
  size_t i = (size_t)blockIdx.x * blockDim.x + threadIdx.x;
  size_t stride = (size_t)gridDim.x * blockDim.x;
  for (; i < n; i += stride) out[i] = (bf16_t)in[i];
}

// Wt[n*1024 + k] = (bf16) W[k*1024 + n]
__global__ __launch_bounds__(256) void transpose_cvt(const float* __restrict__ W,
                                                     bf16_t* __restrict__ Wt) {
  constexpr int E = 1024;
  __shared__ float tile[32][33];
  const int k0 = blockIdx.x * 32, n0 = blockIdx.y * 32;
  const int tx = threadIdx.x & 31, ty = threadIdx.x >> 5;
#pragma unroll
  for (int i = ty; i < 32; i += 8)
    tile[i][tx] = W[(size_t)(k0 + i) * E + (n0 + tx)];
  __syncthreads();
#pragma unroll
  for (int i = ty; i < 32; i += 8)
    Wt[(size_t)(n0 + i) * E + (k0 + tx)] = (bf16_t)tile[tx][i];
}

// ---------------------------------------------------------------------------
// C[4096,1024] = A[4096,1024] * Bt[1024,1024]^T, +bias, *outScale.
// 128x128 block, 8 waves of 32x64; async-LDS double buffered, 1 barrier/step.
// OUT_MODE 0: bf16 head-major [b,h,t,d];  OUT_MODE 1: f32 row-major [M,N].
template <int OUT_MODE>
__global__ __launch_bounds__(256) void gemm_bf16_wmma(const bf16_t* __restrict__ A,
                                                      const bf16_t* __restrict__ Bt,
                                                      const float*  __restrict__ bias,
                                                      void* __restrict__ Cout,
                                                      float outScale) {
  constexpr int TM = 128, TN = 128, TK = 32;
  constexpr int N = 1024, K = 1024;   // E
  __shared__ alignas(16) bf16_t As[2][TM * TK];
  __shared__ alignas(16) bf16_t Bs[2][TN * TK];

  const int m0 = blockIdx.x * TM;
  const int n0 = blockIdx.y * TN;
  const int wave = threadIdx.x >> 5;
  const int wm = wave & 3;   // 4 waves along M
  const int wn = wave >> 2;  // 2 waves along N
  const int lane = threadIdx.x & 31;

  v8f acc[2][4] = {};

  const int row  = threadIdx.x >> 1;        // 0..127
  const int hk16 = (threadIdx.x & 1) * 16;  // 16 bf16 = 32B per thread/matrix

  auto stage = [&](int k0, int buf) {
    const bf16_t* ga = A  + (size_t)(m0 + row) * K + k0 + hk16;
    const bf16_t* gb = Bt + (size_t)(n0 + row) * K + k0 + hk16;
    async_copy_b128(ga,     &As[buf][row * TK + hk16]);
    async_copy_b128(ga + 8, &As[buf][row * TK + hk16 + 8]);
    async_copy_b128(gb,     &Bs[buf][row * TK + hk16]);
    async_copy_b128(gb + 8, &Bs[buf][row * TK + hk16 + 8]);
  };

  stage(0, 0);
  int buf = 0;
  for (int k0 = 0; k0 < K; k0 += TK, buf ^= 1) {
    wait_async();       // my tile-i copies have landed in LDS
    __syncthreads();    // everyone's have
    if (k0 + TK < K) stage(k0 + TK, buf ^ 1);  // prefetch overlaps compute

    const bf16_t* Ab = As[buf];
    const bf16_t* Bb = Bs[buf];
    v16bf af0 = load_frag(Ab + (wm * 32 + 0)  * TK, TK);
    v16bf af1 = load_frag(Ab + (wm * 32 + 16) * TK, TK);
#pragma unroll
    for (int j = 0; j < 4; ++j) {
      v16bf bf = load_frag(Bb + (wn * 64 + j * 16) * TK, TK);
      acc[0][j] = WMMA_BF16(af0, bf, acc[0][j]);
      acc[1][j] = WMMA_BF16(af1, bf, acc[1][j]);
    }
  }

  // Epilogue. C layout: element r -> M = r + 8*(lane>=16), N = lane&15.
  const int nl = lane & 15;
  const int hm = lane >> 4;
#pragma unroll
  for (int i = 0; i < 2; ++i)
#pragma unroll
    for (int j = 0; j < 4; ++j)
#pragma unroll
      for (int r = 0; r < 8; ++r) {
        const int m = m0 + wm * 32 + i * 16 + r + 8 * hm;
        const int n = n0 + wn * 64 + j * 16 + nl;
        const float v = (acc[i][j][r] + bias[n]) * outScale;
        if (OUT_MODE == 0) {
          // T=2048, H=16, d=64 are compile-time: divisions become shifts.
          const int b = m >> 11, t = m & 2047;
          const int h = n >> 6,  d = n & 63;
          ((bf16_t*)Cout)[(((size_t)b * 16 + h) * 2048 + t) * 64 + d] = (bf16_t)v;
        } else {
          ((float*)Cout)[(size_t)m * N + n] = v;
        }
      }
}

// ---------------------------------------------------------------------------
// Flash attention, causal. One workgroup = one (b,h) x 128 q-rows; wave w owns
// q rows [q0, q0+16). K tiles (64x64) via async-LDS DMA, V tiles transposed
// through VGPRs; double-buffered, one barrier per tile. Q arrives pre-scaled
// by 1/sqrt(d). Interior tiles (k0+63 <= q0) take the mask-free path.
__global__ __launch_bounds__(256) void flash_attn_wmma(const bf16_t* __restrict__ Q,
                                                       const bf16_t* __restrict__ Kh,
                                                       const bf16_t* __restrict__ Vh,
                                                       bf16_t* __restrict__ O) {
  constexpr int QT = 128, KT = 64, DH = 64, T = 2048, E = 1024;

  __shared__ alignas(16) bf16_t Ks [2][KT * DH];    // [k][d]      8KB x2
  __shared__ alignas(16) bf16_t Vts[2][DH * KT];    // [d][k]      8KB x2
  __shared__ alignas(16) bf16_t Ps [8 * 16 * KT];   // per-wave P  16KB

  const int bh = blockIdx.x;
  const int qblk = blockIdx.y;
  const int b = bh >> 4, h = bh & 15;  // H=16
  const bf16_t* Qb = Q  + (size_t)bh * T * DH;
  const bf16_t* Kb = Kh + (size_t)bh * T * DH;
  const bf16_t* Vb = Vh + (size_t)bh * T * DH;

  const int wave = threadIdx.x >> 5;
  const int lane = threadIdx.x & 31;
  const int nl = lane & 15;
  const int hm = lane >> 4;
  const int q0 = qblk * QT + wave * 16;
  bf16_t* Pw = Ps + wave * 16 * KT;

  v16bf qf0 = load_frag(Qb + (size_t)q0 * DH + 0,  DH);
  v16bf qf1 = load_frag(Qb + (size_t)q0 * DH + 32, DH);

  float mrow[8], lrow[8];
  v8f o[4] = {};
#pragma unroll
  for (int r = 0; r < 8; ++r) { mrow[r] = -3.0e38f; lrow[r] = 0.0f; }

  const int kr = threadIdx.x >> 2;        // 0..63 : K/V row
  const int cc = (threadIdx.x & 3) * 16;  // 0/16/32/48 : d chunk

  auto stage = [&](int k0, int bufi) {
    // K tile: pure async DMA, 2 x b128 per thread.
    const bf16_t* gk = Kb + (size_t)(k0 + kr) * DH + cc;
    async_copy_b128(gk,     &Ks[bufi][kr * DH + cc]);
    async_copy_b128(gk + 8, &Ks[bufi][kr * DH + cc + 8]);
    // V tile: coalesced b128 global loads, element-transposed ds stores.
    const bf16_t* gv = Vb + (size_t)(k0 + kr) * DH + cc;
    union { uint4 u[2]; bf16_t e[16]; } vv;
    vv.u[0] = *(const uint4*)(gv);
    vv.u[1] = *(const uint4*)(gv + 8);
#pragma unroll
    for (int t = 0; t < 16; ++t)
      Vts[bufi][(cc + t) * KT + kr] = vv.e[t];
  };

  const int kend = qblk * QT + QT;  // causal bound for the whole q-block
  stage(0, 0);
  int bufi = 0;
  for (int k0 = 0; k0 < kend; k0 += KT, bufi ^= 1) {
    wait_async();
    __syncthreads();  // also waits DScnt for the V-transpose stores
    if (k0 + KT < kend) stage(k0 + KT, bufi ^ 1);

    const bf16_t* Kt = Ks[bufi];
    const bf16_t* Vt = Vts[bufi];

    auto tile_body = [&](auto tag) {
      constexpr bool MASKED = decltype(tag)::value;

      // S = Q K^T over 64 columns (4 N-frags x 2 d-steps). Q pre-scaled.
      v8f s[4] = {};
#pragma unroll
      for (int j = 0; j < 4; ++j) {
        s[j] = WMMA_BF16(qf0, load_frag(Kt + (j * 16) * DH + 0,  DH), s[j]);
        s[j] = WMMA_BF16(qf1, load_frag(Kt + (j * 16) * DH + 32, DH), s[j]);
      }

      // Online softmax; row stats reduce across each 16-lane half.
#pragma unroll
      for (int r = 0; r < 8; ++r) {
        const int qi = q0 + r + 8 * hm;
        float e[4];
        float mx = -3.0e38f;
#pragma unroll
        for (int j = 0; j < 4; ++j) {
          e[j] = MASKED ? ((k0 + j * 16 + nl <= qi) ? s[j][r] : -3.0e38f)
                        : s[j][r];
          mx = fmaxf(mx, e[j]);
        }
        mx = red_max16(mx);
        const float mnew = fmaxf(mrow[r], mx);
        const float corr = __expf(mrow[r] - mnew);
        float ps = 0.0f;
#pragma unroll
        for (int j = 0; j < 4; ++j) {
          const float p = __expf(e[j] - mnew);
          ps += p;
          // C-layout -> A-layout via wave-private LDS scratch.
          Pw[(r + 8 * hm) * KT + j * 16 + nl] = (bf16_t)p;
        }
        ps = red_sum16(ps);
        lrow[r] = lrow[r] * corr + ps;
        mrow[r] = mnew;
#pragma unroll
        for (int j = 0; j < 4; ++j) o[j][r] *= corr;
      }
      // Wave-local LDS store->load ordering (split counters).
      asm volatile("s_wait_dscnt 0x0" ::: "memory");

      // O += P * V  (B frag = V^T rows; K-dim 64 = 2 WMMA steps).
      v16bf pf0 = load_frag(Pw + 0,  KT);
      v16bf pf1 = load_frag(Pw + 32, KT);
#pragma unroll
      for (int j = 0; j < 4; ++j) {
        o[j] = WMMA_BF16(pf0, load_frag(Vt + (j * 16) * KT + 0,  KT), o[j]);
        o[j] = WMMA_BF16(pf1, load_frag(Vt + (j * 16) * KT + 32, KT), o[j]);
      }
    };

    // Wave-uniform dispatch: skip fully-masked tiles; mask-free interior.
    if (k0 <= q0 + 15) {
      if (k0 + KT - 1 > q0) tile_body(BoolC<true>{});
      else                  tile_body(BoolC<false>{});
    }
  }

  // Write attention output, bf16 [b, t, h*64 + d].
#pragma unroll
  for (int r = 0; r < 8; ++r) {
    const float inv = 1.0f / lrow[r];
    const int tq = q0 + r + 8 * hm;
    bf16_t* orow = O + ((size_t)b * T + tq) * E + h * DH + nl;
#pragma unroll
    for (int j = 0; j < 4; ++j)
      orow[j * 16] = (bf16_t)(o[j][r] * inv);
  }
}

// ---------------------------------------------------------------------------
extern "C" void kernel_launch(void* const* d_in, const int* in_sizes, int n_in,
                              void* d_out, int out_size, void* d_ws, size_t ws_size,
                              hipStream_t stream) {
  (void)in_sizes; (void)n_in; (void)out_size; (void)ws_size;
  constexpr int Bn = 2, T = 2048, E = 1024, H = 16;
  constexpr size_t M = (size_t)Bn * T;  // 4096

  const float* x  = (const float*)d_in[0];
  const float* y  = (const float*)d_in[1];
  // d_in[2] is the mask == tril -> causality applied analytically.
  const float* Wq = (const float*)d_in[3];
  const float* bq = (const float*)d_in[4];
  const float* Wk = (const float*)d_in[5];
  const float* bk = (const float*)d_in[6];
  const float* Wv = (const float*)d_in[7];
  const float* bv = (const float*)d_in[8];
  const float* Wo = (const float*)d_in[9];
  const float* bo = (const float*)d_in[10];
  float* out = (float*)d_out;

  // Workspace carve-up (~56 MB total).
  char* p = (char*)d_ws;
  auto take = [&](size_t bytes) { char* q = p; p += (bytes + 255) & ~(size_t)255; return q; };
  bf16_t* xb    = (bf16_t*)take(M * E * sizeof(bf16_t));
  bf16_t* yb    = (bf16_t*)take(M * E * sizeof(bf16_t));
  bf16_t* WqT   = (bf16_t*)take((size_t)E * E * sizeof(bf16_t));
  bf16_t* WkT   = (bf16_t*)take((size_t)E * E * sizeof(bf16_t));
  bf16_t* WvT   = (bf16_t*)take((size_t)E * E * sizeof(bf16_t));
  bf16_t* WoT   = (bf16_t*)take((size_t)E * E * sizeof(bf16_t));
  bf16_t* Qh    = (bf16_t*)take(M * E * sizeof(bf16_t));  // [B,H,T,64]
  bf16_t* Kb    = (bf16_t*)take(M * E * sizeof(bf16_t));
  bf16_t* Vb    = (bf16_t*)take(M * E * sizeof(bf16_t));
  bf16_t* attnb = (bf16_t*)take(M * E * sizeof(bf16_t));  // [B,T,E]

  cvt_f32_bf16<<<2048, 256, 0, stream>>>(x, xb, M * E);
  cvt_f32_bf16<<<2048, 256, 0, stream>>>(y, yb, M * E);

  dim3 tg(E / 32, E / 32);
  transpose_cvt<<<tg, 256, 0, stream>>>(Wq, WqT);
  transpose_cvt<<<tg, 256, 0, stream>>>(Wk, WkT);
  transpose_cvt<<<tg, 256, 0, stream>>>(Wv, WvT);
  transpose_cvt<<<tg, 256, 0, stream>>>(Wo, WoT);

  dim3 gg((unsigned)(M / 128), E / 128);
  // 1/sqrt(d) folded into the Q projection.
  gemm_bf16_wmma<0><<<gg, 256, 0, stream>>>(xb, WqT, bq, Qh, 0.125f);
  gemm_bf16_wmma<0><<<gg, 256, 0, stream>>>(yb, WkT, bk, Kb, 1.0f);
  gemm_bf16_wmma<0><<<gg, 256, 0, stream>>>(yb, WvT, bv, Vb, 1.0f);

  flash_attn_wmma<<<dim3(Bn * H, T / 128), 256, 0, stream>>>(Qh, Kb, Vb, attnb);

  gemm_bf16_wmma<1><<<gg, 256, 0, stream>>>(attnb, WoT, bo, out, 1.0f);
}
